// MaskedAttention_17944373363333
// MI455X (gfx1250) — matrix operand order repaired and checked by reading, not compiled, not run on hardware
//
#include <hip/hip_runtime.h>
#include <math.h>

// MaskedAttention for MI455X (gfx1250, wave32, WMMA, TDM).
// B=8, NQ=N=2048, D_IN=D_KEY=D_MODEL=512.
// Reference masks via reshape (not transpose): effective mask for score
// (b,q,n) is mask.flat[(b*2048+q)*2048 + n]; mask applied BEFORE 1/sqrt(512).

typedef __attribute__((ext_vector_type(16))) __bf16          v16bf;
typedef __attribute__((ext_vector_type(8)))  float           v8f;
typedef __attribute__((ext_vector_type(16))) unsigned short  v16u;
typedef __attribute__((ext_vector_type(8)))  unsigned short  v8u;
typedef __attribute__((ext_vector_type(4)))  unsigned int    v4u;
typedef __attribute__((ext_vector_type(8)))  int             v8i;
typedef __attribute__((ext_vector_type(4)))  int             v4i;

__device__ __forceinline__ unsigned short f32_to_bf16_rne(float f) {
  unsigned u = __float_as_uint(f);
  unsigned r = u + 0x7FFFu + ((u >> 16) & 1u);
  return (unsigned short)(r >> 16);
}

__device__ __forceinline__ v8f vzero8() {
  v8f v = {0.f, 0.f, 0.f, 0.f, 0.f, 0.f, 0.f, 0.f};
  return v;
}

__device__ __forceinline__ v8f wmma_bf16(v16u a, v16u b, v8f c) {
  return __builtin_amdgcn_wmma_f32_16x16x32_bf16(
      false, __builtin_bit_cast(v16bf, a),
      false, __builtin_bit_cast(v16bf, b),
      (short)0, c, false, false);
}

__device__ __forceinline__ v16u cat16(v8u lo, v8u hi) {
  return __builtin_shufflevector(lo, hi, 0, 1, 2, 3, 4, 5, 6, 7,
                                 8, 9, 10, 11, 12, 13, 14, 15);
}

// ---------------------------------------------------------------------------
// Tensor Data Mover: 2D bf16 tile load Global -> LDS (ISA ch. 8 D# layout).
// This toolchain exposes the 6-arg builtin:
//   (uint32x4 group0, int32x8 group1, int32x4, int32x4, int32x8, i32 cpol)
// ---------------------------------------------------------------------------
#if __has_builtin(__builtin_amdgcn_tensor_load_to_lds) && \
    __has_builtin(__builtin_amdgcn_s_wait_tensorcnt)
#define HAVE_TDM 1
__device__ __forceinline__ void tdm_load_2d_bf16(
    unsigned lds_addr, const void* gaddr, unsigned tile_d0, unsigned tile_d1,
    unsigned tensor_d0, unsigned tensor_d1, unsigned stride_d0) {
  unsigned long long ga = (unsigned long long)(size_t)gaddr;
  v4u g0;
  g0[0] = 1u;                                          // count=1, user-mode
  g0[1] = lds_addr;                                    // D#.lds_addr [63:32]
  g0[2] = (unsigned)(ga & 0xFFFFFFFFu);                // global_addr[31:0]
  g0[3] = (unsigned)((ga >> 32) & 0x01FFFFFFu) | (2u << 30);  // [56:32]|type=2
  v8i g1;
  g1[0] = (int)(1u << 16);                             // data_size=1 (2B)
  g1[1] = (int)((tensor_d0 & 0xFFFFu) << 16);          // tensor_dim0 lo16
  g1[2] = (int)(((tensor_d0 >> 16) & 0xFFFFu) | ((tensor_d1 & 0xFFFFu) << 16));
  g1[3] = (int)(((tensor_d1 >> 16) & 0xFFFFu) | ((tile_d0 & 0xFFFFu) << 16));
  g1[4] = (int)(tile_d1 & 0xFFFFu);                    // tile_dim1; tile_dim2=0
  g1[5] = (int)stride_d0;                              // dim0_stride lo32
  g1[6] = 0;
  g1[7] = 0;
  v4i z4 = {0, 0, 0, 0};                               // groups 2/3 (2D tensor)
  v8i z8 = {0, 0, 0, 0, 0, 0, 0, 0};
  __builtin_amdgcn_tensor_load_to_lds(g0, g1, z4, z4, z8, 0);
}
#else
#define HAVE_TDM 0
#endif

// ---------------------------------------------------------------------------
// Kernel 0: transpose-convert wq/wk/wv (f32 [k][n]) -> bf16 Wt [n][k]
// ---------------------------------------------------------------------------
__global__ void wt_kernel(const float* __restrict__ wq,
                          const float* __restrict__ wk,
                          const float* __restrict__ wv,
                          unsigned short* __restrict__ wt) {
  int i = blockIdx.x * blockDim.x + threadIdx.x;   // < 3*512*512
  int m = i >> 18;
  int r = i & ((1 << 18) - 1);
  int n = r >> 9;
  int k = r & 511;
  const float* W = (m == 0) ? wq : (m == 1) ? wk : wv;
  wt[i] = f32_to_bf16_rne(W[k * 512 + n]);
}

// ---------------------------------------------------------------------------
// Kernel 1: projections. Block = 8 waves sharing one 16-row input tile
// (staged+converted once in LDS); wave w owns output column tile w.
//   mat 0: Qbf = query @ wq -> bf16 [16384][512]
//   mat 1: Kbf = node  @ wk -> bf16 [16384][512]
//   mat 2: Vt  = (node @ wv)^T per batch -> bf16 [8][512][2048]
// ---------------------------------------------------------------------------
__global__ __launch_bounds__(256) void proj_kernel(
    const float* __restrict__ node, const float* __restrict__ query,
    const unsigned short* __restrict__ wt,
    unsigned short* __restrict__ qbf, unsigned short* __restrict__ kbf,
    unsigned short* __restrict__ vt) {
  __shared__ __align__(16) unsigned short sX[16 * 512];  // 16 KB bf16 tile

  const int tid  = threadIdx.x;
  const int lane = tid & 31;
  const int wave = tid >> 5;
  const int half = lane >> 4;
  const int l15  = lane & 15;
  const int mat  = blockIdx.y;
  const int row0 = (blockIdx.x >> 2) << 4;                    // shared row tile
  const int col0 = (((blockIdx.x & 3) << 3) + wave) << 4;     // per-wave cols

  const float* X = (mat == 0) ? query : node;
  const unsigned short* W = wt + (size_t)mat * (512 * 512);

  // cooperative stage+convert: rows row0..row0+15 are contiguous 8192 floats
  {
    const float* src = X + (size_t)row0 * 512;
#pragma unroll
    for (int i = tid * 4; i < 16 * 512; i += 1024) {
      float4 v = *(const float4*)(src + i);
      unsigned long long pk =
          (unsigned long long)f32_to_bf16_rne(v.x) |
          ((unsigned long long)f32_to_bf16_rne(v.y) << 16) |
          ((unsigned long long)f32_to_bf16_rne(v.z) << 32) |
          ((unsigned long long)f32_to_bf16_rne(v.w) << 48);
      *(unsigned long long*)&sX[i] = pk;
    }
  }
  __syncthreads();

  const unsigned short* wrow = W + (size_t)(col0 + l15) * 512;
  v8f acc = vzero8();
#pragma unroll 4
  for (int kk = 0; kk < 512; kk += 32) {
    const unsigned short* qp = &sX[l15 * 512 + kk + half * 8];
    v16u a = cat16(*(const v8u*)qp, *(const v8u*)(qp + 16));
    v16u b = cat16(*(const v8u*)(wrow + kk + half * 8), *(const v8u*)(wrow + kk + 16 + half * 8));
    acc = wmma_bf16(a, b, acc);
  }

  if (mat == 2) {
#pragma unroll
    for (int r = 0; r < 8; ++r) {
      int g = row0 + r + half * 8;             // = b*2048 + n
      int bi = g >> 11, n = g & 2047;
      vt[((size_t)bi * 512 + col0 + l15) * 2048 + n] = f32_to_bf16_rne(acc[r]);
    }
  } else {
    unsigned short* out = (mat == 0) ? qbf : kbf;
#pragma unroll
    for (int r = 0; r < 8; ++r) {
      int g = row0 + r + half * 8;
      out[(size_t)g * 512 + col0 + l15] = f32_to_bf16_rne(acc[r]);
    }
  }
}

// ---------------------------------------------------------------------------
// Kernel 2: fused flash attention. Block = 8 waves = 32 queries (2 q-tiles),
// full d_model=512. Wave w: score tiles for keys [chunk+16w,+16) (both
// q-tiles, shared K fragment), PV for d in [64w, 64w+64).
// ---------------------------------------------------------------------------
#define CH 128
#define QR 32   // query rows per block

__global__ __launch_bounds__(256) void attn_kernel(
    const unsigned short* __restrict__ qbf, const unsigned short* __restrict__ kbf,
    const unsigned short* __restrict__ vt, const unsigned char* __restrict__ mask,
    float* __restrict__ out) {
  __shared__ __align__(16) unsigned short sQ[QR * 512];   // 32 KB
  __shared__ __align__(16) float sS[QR * CH];             // 16 KB
  __shared__ __align__(16) unsigned short sP[QR * CH];    // 8 KB
  __shared__ float sPart[8 * QR];
  __shared__ float sM[QR], sL[QR], sF[QR];

  const int tid  = threadIdx.x;
  const int lane = tid & 31;
  const int wave = tid >> 5;
  const int half = lane >> 4;
  const int l15  = lane & 15;
  const int q0   = blockIdx.x * QR;
  const int b    = blockIdx.y;

  const unsigned short* Qsrc = qbf + ((size_t)b * 2048 + q0) * 512;
#if HAVE_TDM
  if (wave == 0) {
    // TDM: 2D tile (512 x QR rows) of the [2048 x 512] bf16 Q tensor -> LDS
    tdm_load_2d_bf16((unsigned)(size_t)&sQ[0], Qsrc,
                     /*tile*/ 512, QR, /*tensor*/ 512, 2048, /*stride*/ 512);
    __builtin_amdgcn_s_wait_tensorcnt(0);
  }
#else
  for (int i = tid * 8; i < QR * 512; i += 256 * 8)
    *(v8u*)&sQ[i] = *(const v8u*)&Qsrc[i];
#endif
  if (tid < QR) { sM[tid] = -3.0e38f; sL[tid] = 0.0f; }
  __syncthreads();

  v8f acc[2][4];
#pragma unroll
  for (int qt = 0; qt < 2; ++qt)
#pragma unroll
    for (int t = 0; t < 4; ++t) acc[qt][t] = vzero8();

  const unsigned short* Kb = kbf + (size_t)b * 2048 * 512;
  const unsigned short* Vb = vt + (size_t)b * 512 * 2048;
  const unsigned char*  Mb = mask + ((size_t)b * 2048 + q0) * 2048;
  const float scale = 0.04419417382415922f;  // 1/sqrt(512)

  for (int key0 = 0; key0 < 2048; key0 += CH) {
    // ---- scores: 2 q-tiles x 16 keys per wave, K fragment shared
    v8f sAcc[2] = {vzero8(), vzero8()};
    {
      const unsigned short* krow = Kb + (size_t)(key0 + wave * 16 + l15) * 512;
      if (key0 + CH < 2048)
        __builtin_prefetch(krow + (size_t)CH * 512, 0, 0);
#pragma unroll 2
      for (int kk = 0; kk < 512; kk += 32) {
        v16u bf = cat16(*(const v8u*)(krow + kk + half * 8), *(const v8u*)(krow + kk + 16 + half * 8));
        const unsigned short* qp0 = &sQ[l15 * 512 + kk + half * 8];
        const unsigned short* qp1 = qp0 + 16 * 512;
        sAcc[0] = wmma_bf16(cat16(*(const v8u*)qp0, *(const v8u*)(qp0 + 16)),
                            bf, sAcc[0]);
        sAcc[1] = wmma_bf16(cat16(*(const v8u*)qp1, *(const v8u*)(qp1 + 16)),
                            bf, sAcc[1]);
      }
    }
    // mask (reshape semantics, pre-scale), dump to LDS
#pragma unroll
    for (int qt = 0; qt < 2; ++qt)
#pragma unroll
      for (int r = 0; r < 8; ++r) {
        int row = qt * 16 + r + half * 8;
        int col = wave * 16 + l15;
        unsigned char mk = Mb[(size_t)row * 2048 + key0 + col];
        sS[row * CH + col] = mk ? -3.402823466e38f : sAcc[qt][r];
      }
    __syncthreads();

    // ---- online softmax: 256 threads = 32 rows x 8 segments(16 keys)
    const int row = tid & 31;
    const int seg = tid >> 5;
    float* srow = &sS[row * CH + seg * 16];
    float pmax = srow[0];
#pragma unroll
    for (int j = 1; j < 16; ++j) pmax = fmaxf(pmax, srow[j]);
    sPart[seg * QR + row] = pmax * scale;
    __syncthreads();
    if (tid < QR) {
      float mnew = sM[tid];
#pragma unroll
      for (int sg = 0; sg < 8; ++sg) mnew = fmaxf(mnew, sPart[sg * QR + tid]);
      float f = __expf(sM[tid] - mnew);
      sM[tid] = mnew; sF[tid] = f; sL[tid] *= f;
    }
    __syncthreads();
    {
      float mnew = sM[row];
      float psum = 0.f;
#pragma unroll
      for (int j = 0; j < 16; ++j) {
        float p = __expf(srow[j] * scale - mnew);
        psum += p;
        sP[row * CH + seg * 16 + j] = f32_to_bf16_rne(p);
      }
      sPart[seg * QR + row] = psum;
    }
    __syncthreads();
    if (tid < QR) {
      float cs = 0.f;
#pragma unroll
      for (int sg = 0; sg < 8; ++sg) cs += sPart[sg * QR + tid];
      sL[tid] += cs;
    }
    __syncthreads();

    // ---- rescale accumulators (row = qt*16 + r + 8*half)
#pragma unroll
    for (int qt = 0; qt < 2; ++qt)
#pragma unroll
      for (int r = 0; r < 8; ++r) {
        float fr = sF[qt * 16 + half * 8 + r];
#pragma unroll
        for (int t = 0; t < 4; ++t) acc[qt][t][r] *= fr;
      }

    // ---- P(32xCH) x V(CHx64 slice): V fragment shared across q-tiles
#pragma unroll
    for (int kc = 0; kc < CH; kc += 32) {
      const unsigned short* pp0 = &sP[l15 * CH + kc + half * 8];
      const unsigned short* pp1 = pp0 + 16 * CH;
      v16u a0 = cat16(*(const v8u*)pp0, *(const v8u*)(pp0 + 16));
      v16u a1 = cat16(*(const v8u*)pp1, *(const v8u*)(pp1 + 16));
#pragma unroll
      for (int t = 0; t < 4; ++t) {
        int d = wave * 64 + t * 16 + l15;
        v16u bf = cat16(*(const v8u*)(Vb + (size_t)d * 2048 + key0 + kc + half * 8), *(const v8u*)(Vb + (size_t)d * 2048 + key0 + kc + 16 + half * 8));
        acc[0][t] = wmma_bf16(a0, bf, acc[0][t]);
        acc[1][t] = wmma_bf16(a1, bf, acc[1][t]);
      }
    }
    __syncthreads();  // sS/sP reused next chunk
  }

  // ---- epilogue: divide by row sum, write f32
#pragma unroll
  for (int qt = 0; qt < 2; ++qt)
#pragma unroll
    for (int r = 0; r < 8; ++r) {
      int row = qt * 16 + r + half * 8;
      float rl = 1.0f / sL[row];
      size_t base = ((size_t)b * 2048 + q0 + row) * 512;
#pragma unroll
      for (int t = 0; t < 4; ++t)
        out[base + wave * 64 + t * 16 + l15] = acc[qt][t][r] * rl;
    }
}

// ---------------------------------------------------------------------------
extern "C" void kernel_launch(void* const* d_in, const int* in_sizes, int n_in,
                              void* d_out, int out_size, void* d_ws, size_t ws_size,
                              hipStream_t stream) {
  const float*         node  = (const float*)d_in[0];
  const float*         query = (const float*)d_in[1];
  const unsigned char* mask  = (const unsigned char*)d_in[2];  // jnp bool bytes
  const float*         wq    = (const float*)d_in[3];
  const float*         wk    = (const float*)d_in[4];
  const float*         wv    = (const float*)d_in[5];
  float*               out   = (float*)d_out;

  // Workspace (bf16): Wt[3*512*512] | Qbf[16384*512] | Kbf | Vt  (~49.5 MB)
  unsigned short* wt  = (unsigned short*)d_ws;
  unsigned short* qbf = wt + 3 * 512 * 512;
  unsigned short* kbf = qbf + (size_t)16384 * 512;
  unsigned short* vtb = kbf + (size_t)16384 * 512;

  wt_kernel<<<(3 * 512 * 512) / 256, 256, 0, stream>>>(wq, wk, wv, wt);
  proj_kernel<<<dim3(4096, 3), 256, 0, stream>>>(node, query, wt, qbf, kbf, vtb);
  attn_kernel<<<dim3(64, 8), 256, 0, stream>>>(qbf, kbf, vtb, mask, out);
}
